// EncoderBlock_17025250361729
// MI455X (gfx1250) — matrix operand order repaired and checked by reading, not compiled, not run on hardware
//
#include <hip/hip_runtime.h>
#include <stdint.h>

typedef unsigned int  u32;
typedef unsigned short u16;
typedef __attribute__((ext_vector_type(16))) __bf16 v16bf;
typedef __attribute__((ext_vector_type(8)))  float  v8f;
typedef __attribute__((ext_vector_type(4)))  u32    v4u;

union FragBF { v16bf v; u32 u[8]; v4u q[2]; };

constexpr int kB  = 2;
constexpr int kL  = 2048;
constexpr int kD  = 1024;
constexpr int kH  = 16;
constexpr int kFF = 4096;
constexpr int kM  = kB * kL;        // 4096 tokens

#define USE_ASYNC_LDS 1

__device__ __forceinline__ u16 f2bf(float f) {
  u32 t = __float_as_uint(f);
  t += 0x7FFFu + ((t >> 16) & 1u);      // round-to-nearest-even
  return (u16)(t >> 16);
}
__device__ __forceinline__ u32 pack2(float a, float b) {
  return (u32)f2bf(a) | ((u32)f2bf(b) << 16);
}

#if USE_ASYNC_LDS
__device__ __forceinline__ void async_ld_b64(u32 lds_off, const void* gptr) {
  asm volatile("global_load_async_to_lds_b64 %0, %1, off"
               :: "v"(lds_off), "v"((unsigned long long)(uintptr_t)gptr)
               : "memory");
}
__device__ __forceinline__ void wait_async0() {
  asm volatile("s_wait_asynccnt 0x0" ::: "memory");
}
#endif

// ---------------- f32 -> bf16 weight conversion ----------------
__global__ void conv_bf16(const float* __restrict__ s, u16* __restrict__ d, int n) {
  int i = blockIdx.x * 256 + threadIdx.x;
  if (i < n) d[i] = f2bf(s[i]);
}

// ---------------- LayerNorm over (L,D) per batch ----------------
__global__ void red_partial(const float* __restrict__ x, float* __restrict__ part) {
  const int per = kL * kD;           // 2097152
  const int nChunk = 128;
  const int ch = per / nChunk;       // 16384
  int bidx = blockIdx.y, c = blockIdx.x, tid = threadIdx.x;
  const float* p = x + (size_t)bidx * per + (size_t)c * ch;
  float s = 0.f, q = 0.f;
  for (int i = tid; i < ch; i += 256) { float v = p[i]; s += v; q += v * v; }
  __shared__ float ss[256], sq[256];
  ss[tid] = s; sq[tid] = q; __syncthreads();
  for (int o = 128; o > 0; o >>= 1) {
    if (tid < o) { ss[tid] += ss[tid + o]; sq[tid] += sq[tid + o]; }
    __syncthreads();
  }
  if (tid == 0) {
    part[(bidx * nChunk + c) * 2]     = ss[0];
    part[(bidx * nChunk + c) * 2 + 1] = sq[0];
  }
}

__global__ void red_final(const float* __restrict__ part, float* __restrict__ stats) {
  const int nChunk = 128;
  int bidx = blockIdx.x, tid = threadIdx.x;   // 128 threads
  __shared__ float ss[128], sq[128];
  ss[tid] = part[(bidx * nChunk + tid) * 2];
  sq[tid] = part[(bidx * nChunk + tid) * 2 + 1];
  __syncthreads();
  for (int o = 64; o > 0; o >>= 1) {
    if (tid < o) { ss[tid] += ss[tid + o]; sq[tid] += sq[tid + o]; }
    __syncthreads();
  }
  if (tid == 0) {
    const float invN = 1.f / (float)(kL * kD);
    float mean = ss[0] * invN;
    float var  = sq[0] * invN - mean * mean;
    stats[bidx * 2]     = mean;
    stats[bidx * 2 + 1] = rsqrtf(var + 1e-5f);
  }
}

__global__ void ln_pack(const float* __restrict__ x, const float* __restrict__ stats,
                        u32* __restrict__ outp) {
  const int perPairs = (kL * kD) / 2;   // 1048576 per batch
  int i = blockIdx.x * 256 + threadIdx.x;
  if (i >= kB * perPairs) return;
  int bidx = i / perPairs;
  float mean = stats[bidx * 2], istd = stats[bidx * 2 + 1];
  float a = (x[2 * i]     - mean) * istd;
  float b = (x[2 * i + 1] - mean) * istd;
  outp[i] = pack2(a, b);
}

// ---------------- WMMA GEMM: out[M,N] = A[M,K] @ W[N,K]^T + bias ----------------
// Workgroup tile 256(M) x 64(N); 8 waves, each 32x64 (2 A-frags x 4 B-frags).
// W tile double-buffered in LDS via async VMEM->LDS; A fragments software-
// pipelined one k-step ahead in registers (2x-unrolled loop, nk always even).
// mode 0: bf16 out            mode 1: bf16 out + ReLU
// mode 2: bf16 out scattered to vT[b,h,dh,l]   mode 3: f32 out + resid
__global__ __launch_bounds__(256) void gemm_wmma_k(
    const u16* __restrict__ A, const u16* __restrict__ W, const float* __restrict__ bias,
    u16* __restrict__ outB, float* __restrict__ outF, const float* __restrict__ resid,
    int N, int K, int mode) {
  const int tid = threadIdx.x;
  const int lane = tid & 31, w = tid >> 5;
  const int lm = lane & 15, laneHi = lane >> 4;
  const int wselA = laneHi * 4, bsel = laneHi * 8;
  const int nTile = blockIdx.x * 64;
  const int mBase = blockIdx.y * 256 + w * 32;

  __shared__ u32 sW[2][64 * 18];        // double-buffered, padded stride 18 dwords
  const u32* Wu = (const u32*)W;
  const int Kw2 = K >> 1;
  const u32* Arow0 = (const u32*)A + (size_t)(mBase + lm) * Kw2;
  const u32* Arow1 = (const u32*)A + (size_t)(mBase + 16 + lm) * Kw2;

  // cooperative staging of one 64x32 W tile into sW[buf]
  const int sn = tid >> 3;              // 0..31 (row), +32 for second half
  const int sc = (tid & 7) * 2;         // dword pair column 0..14
#if USE_ASYNC_LDS
  auto stage = [&](int buf, int k0) {
    const int kw = k0 >> 1;
    async_ld_b64((u32)(uintptr_t)&sW[buf][sn * 18 + sc],
                 (const void*)(Wu + (size_t)(nTile + sn) * Kw2 + kw + sc));
    async_ld_b64((u32)(uintptr_t)&sW[buf][(sn + 32) * 18 + sc],
                 (const void*)(Wu + (size_t)(nTile + sn + 32) * Kw2 + kw + sc));
  };
#else
  auto stage = [&](int buf, int k0) {
    const int kw = k0 >> 1;
    sW[buf][sn * 18 + sc]            = Wu[(size_t)(nTile + sn) * Kw2 + kw + sc];
    sW[buf][sn * 18 + sc + 1]        = Wu[(size_t)(nTile + sn) * Kw2 + kw + sc + 1];
    sW[buf][(sn + 32) * 18 + sc]     = Wu[(size_t)(nTile + sn + 32) * Kw2 + kw + sc];
    sW[buf][(sn + 32) * 18 + sc + 1] = Wu[(size_t)(nTile + sn + 32) * Kw2 + kw + sc + 1];
  };
#endif

  auto loadA = [&](FragBF& x0, FragBF& x1, int kw) {
    x0.q[0] = *(const v4u*)(Arow0 + kw + wselA);
    x0.q[1] = *(const v4u*)(Arow0 + kw + 8 + wselA);
    x1.q[0] = *(const v4u*)(Arow1 + kw + wselA);
    x1.q[1] = *(const v4u*)(Arow1 + kw + 8 + wselA);
  };

  v8f zero = {0.f, 0.f, 0.f, 0.f, 0.f, 0.f, 0.f, 0.f};
  v8f acc0[4] = {zero, zero, zero, zero};
  v8f acc1[4] = {zero, zero, zero, zero};

  auto compute = [&](int cur, const FragBF& x0, const FragBF& x1) {
#pragma unroll
    for (int f = 0; f < 4; ++f) {
      FragBF bf;
      int rb = (f * 16 + lm) * 18 + wselA;
#pragma unroll
      for (int j = 0; j < 8; ++j) bf.u[j] = sW[cur][rb + (j & 3) + ((j >> 2) << 3)];
      acc0[f] = __builtin_amdgcn_wmma_f32_16x16x32_bf16(
          false, x0.v, false, bf.v, (short)0, acc0[f], false, false);
      acc1[f] = __builtin_amdgcn_wmma_f32_16x16x32_bf16(
          false, x1.v, false, bf.v, (short)0, acc1[f], false, false);
    }
  };

  const int nk = K >> 5;                // 32 or 128 (always even)
  FragBF aEv0, aEv1, aOd0, aOd1;

  stage(0, 0);
  loadA(aEv0, aEv1, 0);
#if USE_ASYNC_LDS
  wait_async0();
#endif
  __syncthreads();

  for (int kk = 0; kk < nk; kk += 2) {
    // even step: compute on buf0 / aEv while staging buf1 / aOd for kk+1
    stage(1, (kk + 1) << 5);
    loadA(aOd0, aOd1, (kk + 1) << 4);
    compute(0, aEv0, aEv1);
#if USE_ASYNC_LDS
    wait_async0();
#endif
    __syncthreads();
    // odd step: compute on buf1 / aOd while staging buf0 / aEv for kk+2
    if (kk + 2 < nk) {
      stage(0, (kk + 2) << 5);
      loadA(aEv0, aEv1, (kk + 2) << 4);
    }
    compute(1, aOd0, aOd1);
#if USE_ASYNC_LDS
    wait_async0();
#endif
    __syncthreads();
  }

  // store (C/D layout: vgpr r -> row r + 8*laneHi, col = lane&15 within 16-tile)
#pragma unroll
  for (int sub = 0; sub < 2; ++sub) {
#pragma unroll
    for (int f = 0; f < 4; ++f) {
      int n = nTile + f * 16 + lm;
      float bn = bias[n];
#pragma unroll
      for (int r = 0; r < 8; ++r) {
        int m = mBase + sub * 16 + r + laneHi * 8;
        float val = (sub ? acc1[f][r] : acc0[f][r]) + bn;
        if (mode == 1) val = fmaxf(val, 0.f);
        if (mode == 3) {
          size_t idx = (size_t)m * N + n;
          outF[idx] = val + resid[idx];
        } else if (mode == 2) {
          int bidx = m >> 11, l = m & (kL - 1);
          int hh = n >> 6, dh = n & 63;
          outB[(((size_t)bidx * kH + hh) * 64 + dh) * kL + l] = f2bf(val);
        } else {
          outB[(size_t)m * N + n] = f2bf(val);
        }
      }
    }
  }
}

// ---------------- Flash attention: one wave per (b, h, 16-query tile) ----------------
__global__ __launch_bounds__(32) void attn_flash(
    const u16* __restrict__ qb, const u16* __restrict__ kb,
    const u16* __restrict__ vT, u16* __restrict__ ob) {
  const int lane = threadIdx.x;
  const int lm = lane & 15, laneHi = lane >> 4;
  const int wselA = laneHi * 4, bsel = laneHi * 8;
  const int b = blockIdx.z, h = blockIdx.y;
  const int qBase = blockIdx.x << 4;

  __shared__ float sS[16][33];
  __shared__ u32   sPw[16 * 18];
  __shared__ float sAlpha[16];

  // Q fragments (16 queries x 64 dh -> 2 k-steps of 32)
  FragBF qa0, qa1;
  {
    const u32* Qw = (const u32*)(qb + ((size_t)(b * kL + qBase + lm) * kD + h * 64));
    qa0.q[0] = *(const v4u*)(Qw + wselA);
    qa0.q[1] = *(const v4u*)(Qw + 8 + wselA);
    qa1.q[0] = *(const v4u*)(Qw + 16 + wselA);
    qa1.q[1] = *(const v4u*)(Qw + 24 + wselA);
  }

  v8f zero = {0.f, 0.f, 0.f, 0.f, 0.f, 0.f, 0.f, 0.f};
  v8f oacc[4] = {zero, zero, zero, zero};
  float mrow = -3.0e38f, lrow = 0.f;

  for (int k0 = 0; k0 < kL; k0 += 32) {
    // issue V fragment loads first: independent of S, latency hidden behind
    // the S WMMAs and the serial softmax section
    FragBF vb[4];
#pragma unroll
    for (int f = 0; f < 4; ++f) {
      const u32* Vw = (const u32*)(vT + ((((size_t)b * kH + h) * 64 + f * 16 + lm) * kL + k0));
      vb[f].q[0] = *(const v4u*)(Vw + wselA);
      vb[f].q[1] = *(const v4u*)(Vw + 8 + wselA);
    }

    v8f s[2] = {zero, zero};
    // S = Q @ K^T for 32 keys (B-fragment: lane = key column, dword pairs over dh)
#pragma unroll
    for (int ns = 0; ns < 2; ++ns) {
      const u32* Kw = (const u32*)(kb + ((size_t)(b * kL + k0 + ns * 16 + lm) * kD + h * 64));
      FragBF bk0, bk1;
      bk0.q[0] = *(const v4u*)(Kw + wselA);
      bk0.q[1] = *(const v4u*)(Kw + 8 + wselA);
      bk1.q[0] = *(const v4u*)(Kw + 16 + wselA);
      bk1.q[1] = *(const v4u*)(Kw + 24 + wselA);
      s[ns] = __builtin_amdgcn_wmma_f32_16x16x32_bf16(false, qa0.v, false, bk0.v, (short)0, s[ns], false, false);
      s[ns] = __builtin_amdgcn_wmma_f32_16x16x32_bf16(false, qa1.v, false, bk1.v, (short)0, s[ns], false, false);
    }
    // stage scaled S in LDS (scale = 1/sqrt(H) = 0.25, faithful to reference)
#pragma unroll
    for (int ns = 0; ns < 2; ++ns)
#pragma unroll
      for (int r = 0; r < 8; ++r)
        sS[r + laneHi * 8][ns * 16 + lm] = s[ns][r] * 0.25f;
    __syncthreads();

    // online softmax: lanes 0..15 each own one query row
    if (lane < 16) {
      float mx = mrow;
#pragma unroll 8
      for (int c = 0; c < 32; ++c) mx = fmaxf(mx, sS[lane][c]);
      float al = __expf(mrow - mx);
      float sum = 0.f;
      for (int cw = 0; cw < 16; ++cw) {
        float p0 = __expf(sS[lane][2 * cw]     - mx);
        float p1 = __expf(sS[lane][2 * cw + 1] - mx);
        sum += p0 + p1;
        sPw[lane * 18 + cw] = pack2(p0, p1);
      }
      lrow = lrow * al + sum;
      mrow = mx;
      sAlpha[lane] = al;
    }
    __syncthreads();

    // rescale O accumulator rows by alpha
    float av[8];
#pragma unroll
    for (int r = 0; r < 8; ++r) av[r] = sAlpha[r + laneHi * 8];
#pragma unroll
    for (int f = 0; f < 4; ++f)
#pragma unroll
      for (int r = 0; r < 8; ++r) oacc[f][r] *= av[r];

    // P as A-fragment (16x32 bf16) from LDS
    FragBF pa;
    {
      int rb = lm * 18;
#pragma unroll
      for (int j = 0; j < 4; ++j) {
        pa.u[j]     = sPw[rb + wselA + j];
        pa.u[4 + j] = sPw[rb + 8 + wselA + j];
      }
    }
    // O += P @ V  (V fragments already resident)
#pragma unroll
    for (int f = 0; f < 4; ++f)
      oacc[f] = __builtin_amdgcn_wmma_f32_16x16x32_bf16(
          false, pa.v, false, vb[f].v, (short)0, oacc[f], false, false);
    __syncthreads();
  }

  if (lane < 16) sAlpha[lane] = 1.f / lrow;
  __syncthreads();
  float inv[8];
#pragma unroll
  for (int r = 0; r < 8; ++r) inv[r] = sAlpha[r + laneHi * 8];
#pragma unroll
  for (int f = 0; f < 4; ++f)
#pragma unroll
    for (int r = 0; r < 8; ++r) {
      int m = qBase + r + laneHi * 8;
      ob[((size_t)(b * kL + m)) * kD + h * 64 + f * 16 + lm] = f2bf(oacc[f][r] * inv[r]);
    }
}

// ---------------- host-side orchestration ----------------
extern "C" void kernel_launch(void* const* d_in, const int* in_sizes, int n_in,
                              void* d_out, int out_size, void* d_ws, size_t ws_size,
                              hipStream_t stream) {
  const float* x  = (const float*)d_in[0];
  const float* wq = (const float*)d_in[1];
  const float* bq = (const float*)d_in[2];
  const float* wk = (const float*)d_in[3];
  const float* bk = (const float*)d_in[4];
  const float* wv = (const float*)d_in[5];
  const float* bv = (const float*)d_in[6];
  const float* wo = (const float*)d_in[7];
  const float* bo = (const float*)d_in[8];
  const float* w1 = (const float*)d_in[9];
  const float* b1 = (const float*)d_in[10];
  const float* w2 = (const float*)d_in[11];
  const float* b2 = (const float*)d_in[12];
  float* out = (float*)d_out;
  (void)in_sizes; (void)n_in; (void)out_size; (void)ws_size;

  char* base = (char*)d_ws;
  size_t off = 0;
  auto take = [&](size_t bytes) -> void* {
    void* p = base + off;
    off = (off + bytes + 255) & ~(size_t)255;
    return p;
  };
  u16* wqb = (u16*)take((size_t)kD * kD * 2);
  u16* wkb = (u16*)take((size_t)kD * kD * 2);
  u16* wvb = (u16*)take((size_t)kD * kD * 2);
  u16* wob = (u16*)take((size_t)kD * kD * 2);
  u16* w1b = (u16*)take((size_t)kFF * kD * 2);
  u16* w2b = (u16*)take((size_t)kD * kFF * 2);
  u16* hb  = (u16*)take((size_t)kM * kD * 2);
  u16* qbf = (u16*)take((size_t)kM * kD * 2);
  u16* kbf = (u16*)take((size_t)kM * kD * 2);
  u16* vT  = (u16*)take((size_t)kM * kD * 2);
  u16* obf = (u16*)take((size_t)kM * kD * 2);
  float* ubuf = (float*)take((size_t)kM * kD * 4);
  u16* yb  = (u16*)take((size_t)kM * kD * 2);
  u16* fb  = (u16*)take((size_t)kM * kFF * 2);
  float* part   = (float*)take((size_t)kB * 128 * 2 * 4);
  float* stats1 = (float*)take(256);
  float* stats2 = (float*)take(256);

  // 1. weights -> bf16
  conv_bf16<<<(kD * kD + 255) / 256, 256, 0, stream>>>(wq, wqb, kD * kD);
  conv_bf16<<<(kD * kD + 255) / 256, 256, 0, stream>>>(wk, wkb, kD * kD);
  conv_bf16<<<(kD * kD + 255) / 256, 256, 0, stream>>>(wv, wvb, kD * kD);
  conv_bf16<<<(kD * kD + 255) / 256, 256, 0, stream>>>(wo, wob, kD * kD);
  conv_bf16<<<(kFF * kD + 255) / 256, 256, 0, stream>>>(w1, w1b, kFF * kD);
  conv_bf16<<<(kD * kFF + 255) / 256, 256, 0, stream>>>(w2, w2b, kD * kFF);

  // 2. LN1 -> hb (bf16)
  red_partial<<<dim3(128, kB), 256, 0, stream>>>(x, part);
  red_final<<<kB, 128, 0, stream>>>(part, stats1);
  ln_pack<<<(kB * kL * kD / 2 + 255) / 256, 256, 0, stream>>>(x, stats1, (u32*)hb);

  // 3. QKV projections (V written transposed as vT[b,h,dh,l])
  dim3 gD(kD / 64, kM / 256);
  gemm_wmma_k<<<gD, 256, 0, stream>>>(hb, wqb, bq, qbf, nullptr, nullptr, kD, kD, 0);
  gemm_wmma_k<<<gD, 256, 0, stream>>>(hb, wkb, bk, kbf, nullptr, nullptr, kD, kD, 0);
  gemm_wmma_k<<<gD, 256, 0, stream>>>(hb, wvb, bv, vT,  nullptr, nullptr, kD, kD, 2);

  // 4. attention
  attn_flash<<<dim3(kL / 16, kH, kB), 32, 0, stream>>>(qbf, kbf, vT, obf);

  // 5. O-proj + residual x -> ubuf (f32)
  gemm_wmma_k<<<gD, 256, 0, stream>>>(obf, wob, bo, nullptr, ubuf, x, kD, kD, 3);

  // 6. LN2 -> yb (bf16)
  red_partial<<<dim3(128, kB), 256, 0, stream>>>(ubuf, part);
  red_final<<<kB, 128, 0, stream>>>(part, stats2);
  ln_pack<<<(kB * kL * kD / 2 + 255) / 256, 256, 0, stream>>>(ubuf, stats2, (u32*)yb);

  // 7. FFN1 (ReLU) -> fb (bf16)
  gemm_wmma_k<<<dim3(kFF / 64, kM / 256), 256, 0, stream>>>(yb, w1b, b1, fb, nullptr, nullptr, kFF, kD, 1);

  // 8. FFN2 + original-x residual -> d_out (f32)
  gemm_wmma_k<<<gD, 256, 0, stream>>>(fb, w2b, b2, nullptr, out, x, kD, kFF, 3);
}
